// LinearAttention_103079215338
// MI455X (gfx1250) — matrix operand order repaired and checked by reading, not compiled, not run on hardware
//
#include <hip/hip_runtime.h>

// ---------------------------------------------------------------------------
// LinearAttention for MI455X (gfx1250): bf16 WMMA pipeline, round 3.
//   B=4, N=4096, D=1024.  ~206 GFLOP vs ~0.45 GB HBM.
// Round-3 changes vs round 2:
//   * LDS staging now uses gfx1250 async-tensor path:
//       global_load_async_to_lds_b128 (ASYNCcnt) + s_wait_asynccnt, double-
//       buffered LDS -> copies for tile k+1 overlap the WMMA burst on tile k,
//       one barrier per tile, no VGPR round-trip.
//   * kv kernel emits KV^T so the final GEMM's B operand is async-stageable.
// Workspace use: 144 MB (xh 32 + Wt 8 + Q 32 + Kt 32 + Vt 32 + KVt 8).
// ---------------------------------------------------------------------------

typedef __bf16 bf16;
typedef __attribute__((ext_vector_type(16))) __bf16 v16bf;
typedef __attribute__((ext_vector_type(8)))  __bf16 v8bf;
typedef __attribute__((ext_vector_type(4)))  __bf16 v4bf;
typedef __attribute__((ext_vector_type(8)))  float  v8f;
typedef __attribute__((ext_vector_type(4)))  float  v4f;

#define D_DIM 1024
#define N_TOK 4096
#define B_SZ  4
#define M_TOT (B_SZ * N_TOK)   // 16384
#define LPITCH 48              // LDS row pitch in bf16 (96B: 16B aligned, conflict-breaking)
#define TPITCH 144             // LDS pitch for 32x128 transpose restage (288B, 16B aligned)

__device__ __forceinline__ bf16 f2bf(float f) {
  union { float f; unsigned u; } in; in.f = f;
  unsigned u = in.u;
  u += 0x7fffu + ((u >> 16) & 1u);          // round-to-nearest-even
  union { unsigned short s; bf16 b; } out; out.s = (unsigned short)(u >> 16);
  return out.b;
}

__device__ __forceinline__ v16bf mk16(v8bf lo, v8bf hi) {
  v16bf r;
#pragma unroll
  for (int i = 0; i < 8; i++) { r[i] = lo[i]; r[i + 8] = hi[i]; }
  return r;
}

// A fragment (16x32 bf16): lane l holds row m=l%16, K = (l/16)*8 + {0..7, 16..23}
__device__ __forceinline__ v16bf frag_a(const bf16* tile, int lane) {
  const bf16* p = tile + (lane & 15) * LPITCH + ((lane >> 4) * 8);
  return mk16(*(const v8bf*)(p), *(const v8bf*)(p + 16));
}

// B fragment (32x16 bf16): lane l holds col n=l%16, K = (l/16)*8 + {0..7, 16..23}
__device__ __forceinline__ v16bf frag_b(const bf16* tile, int lane) {
  const bf16* p = tile + (lane & 15) * LPITCH + ((lane >> 4) * 8);
  return mk16(*(const v8bf*)(p), *(const v8bf*)(p + 16));
}

__device__ __forceinline__ v8f wmma_bf16(v16bf a, v16bf b, v8f c) {
  return __builtin_amdgcn_wmma_f32_16x16x32_bf16(false, a, false, b, (short)0, c,
                                                 false, false);
}

// ---- gfx1250 async global->LDS copy (ASYNCcnt), 16B per lane --------------
__device__ __forceinline__ void async_cp16(bf16* lds, const bf16* g) {
  unsigned l = (unsigned)(unsigned long long)lds;  // generic LDS addr[31:0] == LDS offset
  asm volatile("global_load_async_to_lds_b128 %0, %1, off"
               :: "v"(l), "v"(g) : "memory");
}
__device__ __forceinline__ void async_wait0() {
  asm volatile("s_wait_asynccnt 0x0" ::: "memory");
}

// ---------------------------------------------------------------- cast x ---
__global__ __launch_bounds__(256) void cast_f32_bf16_kernel(
    const float* __restrict__ src, bf16* __restrict__ dst, int n4) {
  int i = blockIdx.x * 256 + threadIdx.x;
  if (i < n4) {
    v4f v = *(const v4f*)(src + (size_t)i * 4);
    v4bf o;
#pragma unroll
    for (int j = 0; j < 4; j++) o[j] = f2bf(v[j]);
    *(v4bf*)(dst + (size_t)i * 4) = o;
  }
}

// ------------------------------------------- cast+transpose a DxD weight ---
__global__ __launch_bounds__(256) void weight_t_kernel(
    const float* __restrict__ W, bf16* __restrict__ Wt) {
  __shared__ float tile[32][33];
  int t = threadIdx.x;
  int r0 = blockIdx.x * 32, c0 = blockIdx.y * 32;
  {
    int r = t >> 3, c = (t & 7) * 4;
    v4f v = *(const v4f*)(W + (size_t)(r0 + r) * D_DIM + c0 + c);
#pragma unroll
    for (int i = 0; i < 4; i++) tile[r][c + i] = v[i];
  }
  __syncthreads();
  {
    int cc = t >> 3, rr = (t & 7) * 4;
    v4bf o;
#pragma unroll
    for (int i = 0; i < 4; i++) o[i] = f2bf(tile[rr + i][cc]);
    *(v4bf*)(Wt + (size_t)(c0 + cc) * D_DIM + r0 + rr) = o;
  }
}

// ------------------------------------------------------- fused projection ---
// X[16384,1024] x {Wql,Wqr,Wk,Wv} -> Q (gated), Kt/Vt ([b][d][tok]).
__global__ __launch_bounds__(256) void proj_kernel(
    const bf16* __restrict__ xh,
    const bf16* __restrict__ wt0, const bf16* __restrict__ wt1,
    const bf16* __restrict__ wt2, const bf16* __restrict__ wt3,
    bf16* __restrict__ Q, bf16* __restrict__ Kt, bf16* __restrict__ Vt) {
  __shared__ bf16 As[2][128 * LPITCH];       // 24 KB
  __shared__ bf16 Bs[2][4][32 * LPITCH];     // 24 KB

  int t = threadIdx.x, wave = t >> 5, lane = t & 31;
  int m0 = blockIdx.x * 128;
  int n0 = blockIdx.y * 32;

  v8f acc[4][2];
#pragma unroll
  for (int w = 0; w < 4; w++)
#pragma unroll
    for (int j = 0; j < 2; j++) acc[w][j] = (v8f)(0.0f);

  // A-stage: 2 x 16B per thread;  B-stage: 2 x 16B per thread (one weight)
  int ar = t >> 1, acs = (t & 1) * 16;
  int bw = t >> 6, wr = (t & 63) >> 1, wcs = (t & 1) * 16;
  const bf16* wp = wt0;
  wp = (bw == 1) ? wt1 : wp;
  wp = (bw == 2) ? wt2 : wp;
  wp = (bw == 3) ? wt3 : wp;

  const bf16* aptr = xh + (size_t)(m0 + ar) * D_DIM + acs;
  const bf16* wptr = wp + (size_t)(n0 + wr) * D_DIM + wcs;
  bf16* bsrow[2] = {&Bs[0][bw][wr * LPITCH + wcs], &Bs[1][bw][wr * LPITCH + wcs]};

  auto stage = [&](int buf, int k) {
    async_cp16(&As[buf][ar * LPITCH + acs],     aptr + k);
    async_cp16(&As[buf][ar * LPITCH + acs + 8], aptr + k + 8);
    async_cp16(bsrow[buf],     wptr + k);
    async_cp16(bsrow[buf] + 8, wptr + k + 8);
  };
  auto compute = [&](int buf) {
    v16bf a = frag_a(&As[buf][wave * 16 * LPITCH], lane);
    v16bf bfr[4][2];
#pragma unroll
    for (int w = 0; w < 4; w++)
#pragma unroll
      for (int j = 0; j < 2; j++)
        bfr[w][j] = frag_b(&Bs[buf][w][j * 16 * LPITCH], lane);
#pragma unroll
    for (int w = 0; w < 4; w++)
#pragma unroll
      for (int j = 0; j < 2; j++)
        acc[w][j] = wmma_bf16(a, bfr[w][j], acc[w][j]);
  };

  stage(0, 0);
  for (int k0 = 0; k0 < D_DIM; k0 += 64) {
    async_wait0();
    __syncthreads();
    if (k0 + 32 < D_DIM) stage(1, k0 + 32);
    compute(0);
    async_wait0();
    __syncthreads();
    if (k0 + 64 < D_DIM) stage(0, k0 + 64);
    compute(1);
  }

  // ---------------- epilogue: LDS restage -> coalesced b128 stores ----------
  int lh = lane >> 4, lc = lane & 15;
  int bidx = m0 >> 12;
  int tok0 = m0 & (N_TOK - 1);
  bf16* Qs = &As[0][0];                      // 128 x 32, pitch 32
  bf16* Ts = &Bs[0][0][0];                   // 32 x 128, pitch TPITCH

  __syncthreads();
#pragma unroll
  for (int j = 0; j < 2; j++)
#pragma unroll
    for (int r = 0; r < 8; r++)
      Qs[(wave * 16 + r + 8 * lh) * 32 + j * 16 + lc] =
          f2bf(acc[0][j][r] * acc[1][j][r]);
  __syncthreads();
  {
    int r = t >> 1, cs = (t & 1) * 16;
    bf16* dst = Q + (size_t)(m0 + r) * D_DIM + n0 + cs;
    *(v8bf*)(dst)     = *(const v8bf*)&Qs[r * 32 + cs];
    *(v8bf*)(dst + 8) = *(const v8bf*)&Qs[r * 32 + cs + 8];
  }

  __syncthreads();
#pragma unroll
  for (int j = 0; j < 2; j++)
#pragma unroll
    for (int r = 0; r < 8; r++)
      Ts[(j * 16 + lc) * TPITCH + wave * 16 + r + 8 * lh] = f2bf(acc[2][j][r]);
  __syncthreads();
  {
    int r = t >> 3, cs = (t & 7) * 16;
    bf16* dst = Kt + ((size_t)(bidx * D_DIM + n0 + r)) * N_TOK + tok0 + cs;
    *(v8bf*)(dst)     = *(const v8bf*)&Ts[r * TPITCH + cs];
    *(v8bf*)(dst + 8) = *(const v8bf*)&Ts[r * TPITCH + cs + 8];
  }

  __syncthreads();
#pragma unroll
  for (int j = 0; j < 2; j++)
#pragma unroll
    for (int r = 0; r < 8; r++)
      Ts[(j * 16 + lc) * TPITCH + wave * 16 + r + 8 * lh] = f2bf(acc[3][j][r]);
  __syncthreads();
  {
    int r = t >> 3, cs = (t & 7) * 16;
    bf16* dst = Vt + ((size_t)(bidx * D_DIM + n0 + r)) * N_TOK + tok0 + cs;
    *(v8bf*)(dst)     = *(const v8bf*)&Ts[r * TPITCH + cs];
    *(v8bf*)(dst + 8) = *(const v8bf*)&Ts[r * TPITCH + cs + 8];
  }
}

// ----------------------------------------------- kv^T: KVt[b][e][d] --------
// A = Kt [1024 d x 4096 tok]; B staged from Vt rows; epilogue writes KV^T.
__global__ __launch_bounds__(256) void kv_kernel(
    const bf16* __restrict__ Kt, const bf16* __restrict__ Vt,
    bf16* __restrict__ KVt) {
  __shared__ bf16 As[2][128 * LPITCH];       // 24 KB
  __shared__ bf16 Bs[2][32 * LPITCH];        //  6 KB
  int t = threadIdx.x, wave = t >> 5, lane = t & 31;
  int b = blockIdx.z;
  int m0 = blockIdx.x * 128;  // d
  int n0 = blockIdx.y * 32;   // e

  v8f acc[2] = {(v8f)(0.0f), (v8f)(0.0f)};
  int ar = t >> 1, acs = (t & 1) * 16;
  int br = t >> 2, bcs = (t & 3) * 8;        // 128 threads cover 32x32 B tile

  const bf16* aptr = Kt + (size_t)b * D_DIM * N_TOK + (size_t)(m0 + ar) * N_TOK + acs;
  const bf16* bptr = Vt + (size_t)b * D_DIM * N_TOK + (size_t)(n0 + br) * N_TOK + bcs;

  auto stage = [&](int buf, int k) {
    async_cp16(&As[buf][ar * LPITCH + acs],     aptr + k);
    async_cp16(&As[buf][ar * LPITCH + acs + 8], aptr + k + 8);
    if (t < 128) async_cp16(&Bs[buf][br * LPITCH + bcs], bptr + k);
  };
  auto compute = [&](int buf) {
    v16bf a  = frag_a(&As[buf][wave * 16 * LPITCH], lane);
    v16bf b0 = frag_b(&Bs[buf][0 * 16 * LPITCH], lane);
    v16bf b1 = frag_b(&Bs[buf][1 * 16 * LPITCH], lane);
    acc[0] = wmma_bf16(a, b0, acc[0]);
    acc[1] = wmma_bf16(a, b1, acc[1]);
  };

  stage(0, 0);
  for (int k0 = 0; k0 < N_TOK; k0 += 64) {
    async_wait0();
    __syncthreads();
    if (k0 + 32 < N_TOK) stage(1, k0 + 32);
    compute(0);
    async_wait0();
    __syncthreads();
    if (k0 + 64 < N_TOK) stage(0, k0 + 64);
    compute(1);
  }

  // transpose restage -> KVt[b][e][d], coalesced b128 rows
  int lh = lane >> 4, lc = lane & 15;
  bf16* Ts = &As[0][0];                      // 32 x 128, pitch TPITCH
  __syncthreads();
#pragma unroll
  for (int j = 0; j < 2; j++)
#pragma unroll
    for (int r = 0; r < 8; r++)
      Ts[(j * 16 + lc) * TPITCH + wave * 16 + r + 8 * lh] = f2bf(acc[j][r]);
  __syncthreads();
  {
    int r = t >> 3, cs = (t & 7) * 16;
    bf16* dst = KVt + (size_t)b * D_DIM * D_DIM + (size_t)(n0 + r) * D_DIM + m0 + cs;
    *(v8bf*)(dst)     = *(const v8bf*)&Ts[r * TPITCH + cs];
    *(v8bf*)(dst + 8) = *(const v8bf*)&Ts[r * TPITCH + cs + 8];
  }
}

// ----------------------------------------------------------- out = Q @ kv ---
// A = Q[b] row-major; B = KVt[b][e][d] rows -> async-stageable, contiguous.
__global__ __launch_bounds__(256) void attn_out_kernel(
    const bf16* __restrict__ Q, const bf16* __restrict__ KVt,
    float* __restrict__ out) {
  __shared__ bf16 As[2][128 * LPITCH];
  __shared__ bf16 Bs[2][32 * LPITCH];
  int t = threadIdx.x, wave = t >> 5, lane = t & 31;
  int b = blockIdx.z;
  int m0 = blockIdx.x * 128;  // token
  int n0 = blockIdx.y * 32;   // e

  v8f acc[2] = {(v8f)(0.0f), (v8f)(0.0f)};
  int ar = t >> 1, acs = (t & 1) * 16;
  int br = t >> 2, bcs = (t & 3) * 8;

  const bf16* aptr = Q + (size_t)b * N_TOK * D_DIM + (size_t)(m0 + ar) * D_DIM + acs;
  const bf16* bptr = KVt + (size_t)b * D_DIM * D_DIM + (size_t)(n0 + br) * D_DIM + bcs;

  auto stage = [&](int buf, int k) {
    async_cp16(&As[buf][ar * LPITCH + acs],     aptr + k);
    async_cp16(&As[buf][ar * LPITCH + acs + 8], aptr + k + 8);
    if (t < 128) async_cp16(&Bs[buf][br * LPITCH + bcs], bptr + k);
  };
  auto compute = [&](int buf) {
    v16bf a  = frag_a(&As[buf][wave * 16 * LPITCH], lane);
    v16bf b0 = frag_b(&Bs[buf][0 * 16 * LPITCH], lane);
    v16bf b1 = frag_b(&Bs[buf][1 * 16 * LPITCH], lane);
    acc[0] = wmma_bf16(a, b0, acc[0]);
    acc[1] = wmma_bf16(a, b1, acc[1]);
  };

  stage(0, 0);
  for (int k0 = 0; k0 < D_DIM; k0 += 64) {
    async_wait0();
    __syncthreads();
    if (k0 + 32 < D_DIM) stage(1, k0 + 32);
    compute(0);
    async_wait0();
    __syncthreads();
    if (k0 + 64 < D_DIM) stage(0, k0 + 64);
    compute(1);
  }

  // f32 output: per VGPR, each half-wave writes a full contiguous 64B line.
  int mg0 = m0 + wave * 16;
  int lh = lane >> 4, lc = lane & 15;
#pragma unroll
  for (int j = 0; j < 2; j++)
#pragma unroll
    for (int r = 0; r < 8; r++) {
      int m = mg0 + r + 8 * lh;
      int n = n0 + j * 16 + lc;
      out[(size_t)b * N_TOK * D_DIM + (size_t)m * D_DIM + n] = acc[j][r];
    }
}

// ---------------------------------------------------------------------------
extern "C" void kernel_launch(void* const* d_in, const int* in_sizes, int n_in,
                              void* d_out, int out_size, void* d_ws, size_t ws_size,
                              hipStream_t stream) {
  const float* x   = (const float*)d_in[0];
  const float* wql = (const float*)d_in[1];
  const float* wqr = (const float*)d_in[2];
  const float* wk  = (const float*)d_in[3];
  const float* wv  = (const float*)d_in[4];
  float* out = (float*)d_out;

  char* ws = (char*)d_ws;
  size_t off = 0;
  auto alloc = [&](size_t bytes) { void* p = ws + off; off += (bytes + 255) & ~(size_t)255; return p; };

  bf16* xh = (bf16*)alloc((size_t)M_TOT * D_DIM * 2);          // 32 MB
  bf16* wt[4];
  for (int i = 0; i < 4; i++) wt[i] = (bf16*)alloc((size_t)D_DIM * D_DIM * 2);  // 8 MB
  bf16* Q   = (bf16*)alloc((size_t)M_TOT * D_DIM * 2);         // 32 MB
  bf16* Kt  = (bf16*)alloc((size_t)B_SZ * D_DIM * N_TOK * 2);  // 32 MB
  bf16* Vt  = (bf16*)alloc((size_t)B_SZ * D_DIM * N_TOK * 2);  // 32 MB
  bf16* KVt = (bf16*)alloc((size_t)B_SZ * D_DIM * D_DIM * 2);  // 8 MB
  (void)ws_size; (void)in_sizes; (void)n_in; (void)out_size;

  int n4 = (M_TOT * D_DIM) / 4;
  cast_f32_bf16_kernel<<<(n4 + 255) / 256, 256, 0, stream>>>(x, xh, n4);
  dim3 wg(D_DIM / 32, D_DIM / 32);
  weight_t_kernel<<<wg, 256, 0, stream>>>(wql, wt[0]);
  weight_t_kernel<<<wg, 256, 0, stream>>>(wqr, wt[1]);
  weight_t_kernel<<<wg, 256, 0, stream>>>(wk,  wt[2]);
  weight_t_kernel<<<wg, 256, 0, stream>>>(wv,  wt[3]);

  dim3 pg(M_TOT / 128, D_DIM / 32);
  proj_kernel<<<pg, 256, 0, stream>>>(xh, wt[0], wt[1], wt[2], wt[3], Q, Kt, Vt);

  dim3 kvg(D_DIM / 128, D_DIM / 32, B_SZ);
  kv_kernel<<<kvg, 256, 0, stream>>>(Kt, Vt, KVt);

  dim3 og(N_TOK / 128, D_DIM / 32, B_SZ);
  attn_out_kernel<<<og, 256, 0, stream>>>(Q, KVt, out);
}
